// SplitFC_37709812859643
// MI455X (gfx1250) — matrix operand order repaired and checked by reading, not compiled, not run on hardware
//
#include <hip/hip_runtime.h>
#include <hip/hip_bf16.h>

// SplitFC on MI455X: out = x @ fixed_quant(W)^T + sum(fixed_quant(b))
//   One GEMM: M=1024, N=4096, K=4096 (K = nb*128 + i).
// quantized weights k/128 are EXACT in bf16; x split into hi+lo bf16 ->
// two v_wmma_f32_16x16x32_bf16 per tile, fp32 accumulate.
// Fast path: preprocess W->bf16 (K-major) and x->(hi,lo) bf16 once, then the
// GEMM stages raw bf16 tiles with GLOBAL_LOAD_ASYNC_TO_LDS_B128 (ASYNCcnt).
// Fallback (ws too small): round-1 fused conversion kernel.

typedef __bf16   v16bf __attribute__((ext_vector_type(16)));
typedef __bf16   v8bf  __attribute__((ext_vector_type(8)));
typedef float    v8f   __attribute__((ext_vector_type(8)));
typedef int      v4i   __attribute__((vector_size(16)));   // matches builtin proto

#define M_DIM   1024
#define K_DIM   4096
#define N_DIM   4096
#define NBLK    32
#define IN_BLK  128
#define MT      128
#define NT      128
#define KC      32
#define NKC     (K_DIM / KC)   // 128
#define LDAP    40             // padded LDS row (80B = 20 banks, conflict-free)

#if defined(__has_builtin)
# if __has_builtin(__builtin_amdgcn_global_load_async_to_lds_b128)
#  define HAVE_ASYNC 1
# endif
#endif
#ifndef HAVE_ASYNC
# define HAVE_ASYNC 0
#endif

#define AS1 __attribute__((address_space(1)))
#define AS3 __attribute__((address_space(3)))

__device__ __forceinline__ float fixed_quant_dev(float w) {
    float q = rintf(w * 128.0f);            // RNE, matches jnp.round
    q = fmaxf(q, -128.0f);
    q = fminf(q, 127.0f);
    return q * 0.0078125f;                  // /128 ; exact in bf16
}

__device__ __forceinline__ unsigned bf16bits(float f) {
    return __float_as_uint(f) >> 16;        // truncation; exact for quantized vals
}

__device__ __forceinline__ void wait_async0() {
#if __has_builtin(__builtin_amdgcn_s_wait_asynccnt)
    __builtin_amdgcn_s_wait_asynccnt(0);
#else
    asm volatile("s_wait_asynccnt 0x0" ::: "memory");
#endif
}

__device__ __forceinline__ void async_cp16(const __bf16* g, __bf16* l) {
#if HAVE_ASYNC
    __builtin_amdgcn_global_load_async_to_lds_b128(
        (AS1 v4i*)g, (AS3 v4i*)l, 0, 0);
#endif
}

// ---------------------------------------------------------------- bias sum
__global__ void bias_sum_kernel(const float* __restrict__ b,
                                float* __restrict__ bsum) {
    int o = blockIdx.x * blockDim.x + threadIdx.x;
    if (o < N_DIM) {
        float s = 0.0f;
        #pragma unroll 8
        for (int nb = 0; nb < NBLK; ++nb) s += fixed_quant_dev(b[nb * N_DIM + o]);
        bsum[o] = s;
    }
}

// ------------------------------------------------- preprocess: W -> bf16 K-major
__global__ __launch_bounds__(256) void quantW_kernel(const float* __restrict__ W,
                                                     __bf16* __restrict__ WQ) {
    size_t gid = (size_t)blockIdx.x * 256 + threadIdx.x;   // one 8-elem segment
    int k8 = (int)(gid & 511);          // 512 segments per output row
    int o  = (int)(gid >> 9);           // 0..4095
    int nb = k8 >> 4;                   // (k8*8)/128
    int i  = (k8 << 3) & 127;
    const float* src = W + ((size_t)nb * N_DIM + o) * IN_BLK + i;
    float4 a = ((const float4*)src)[0];
    float4 b = ((const float4*)src)[1];
    uint4 p;
    p.x = bf16bits(fixed_quant_dev(a.x)) | (bf16bits(fixed_quant_dev(a.y)) << 16);
    p.y = bf16bits(fixed_quant_dev(a.z)) | (bf16bits(fixed_quant_dev(a.w)) << 16);
    p.z = bf16bits(fixed_quant_dev(b.x)) | (bf16bits(fixed_quant_dev(b.y)) << 16);
    p.w = bf16bits(fixed_quant_dev(b.z)) | (bf16bits(fixed_quant_dev(b.w)) << 16);
    *(uint4*)(WQ + (size_t)o * K_DIM + (k8 << 3)) = p;
}

// ------------------------------------------------- preprocess: x -> hi/lo bf16
__global__ __launch_bounds__(256) void splitX_kernel(const float* __restrict__ x,
                                                     __bf16* __restrict__ XH,
                                                     __bf16* __restrict__ XL) {
    size_t off = ((size_t)blockIdx.x * 256 + threadIdx.x) * 8;
    float4 a = ((const float4*)(x + off))[0];
    float4 b = ((const float4*)(x + off))[1];
    float f[8] = {a.x, a.y, a.z, a.w, b.x, b.y, b.z, b.w};
    unsigned h[8]; float l[8];
    #pragma unroll
    for (int j = 0; j < 8; ++j) {
        h[j] = __float_as_uint(f[j]) & 0xFFFF0000u;
        l[j] = f[j] - __uint_as_float(h[j]);       // exact residual
    }
    uint4 hp, lp;
    hp.x = (h[0] >> 16) | h[1];  hp.y = (h[2] >> 16) | h[3];
    hp.z = (h[4] >> 16) | h[5];  hp.w = (h[6] >> 16) | h[7];
    lp.x = bf16bits(l[0]) | (__float_as_uint(l[1]) & 0xFFFF0000u);
    lp.y = bf16bits(l[2]) | (__float_as_uint(l[3]) & 0xFFFF0000u);
    lp.z = bf16bits(l[4]) | (__float_as_uint(l[5]) & 0xFFFF0000u);
    lp.w = bf16bits(l[6]) | (__float_as_uint(l[7]) & 0xFFFF0000u);
    *(uint4*)(XH + off) = hp;
    *(uint4*)(XL + off) = lp;
}

// ---------------------------------------------------------- fast GEMM (bf16 in)
__global__ __launch_bounds__(256) void splitfc_gemm_pre_kernel(
    const __bf16* __restrict__ XH, const __bf16* __restrict__ XL,
    const __bf16* __restrict__ WQ, const float* __restrict__ bsum,
    float* __restrict__ out) {

    __shared__ __bf16 ldsAhi[2][MT * LDAP];
    __shared__ __bf16 ldsAlo[2][MT * LDAP];
    __shared__ __bf16 ldsB  [2][NT * LDAP];

    const int tid  = (int)threadIdx.x;
    const int lane = tid & 31;
    const int wave = tid >> 5;
    const int wm   = wave >> 1;         // 0..3
    const int wn   = wave & 1;          // 0..1
    const int lg   = lane >> 4;
    const int l15  = lane & 15;
    const int m0   = (int)blockIdx.x * MT;   // gridDim.x = 8
    const int n0   = (int)blockIdx.y * NT;   // gridDim.y = 32

    // staging map: idx = q*256+tid ; 4 x 16B segments per 64B row, 128 rows
    size_t aoff[2], boff[2];
    int    loff[2];
    #pragma unroll
    for (int q = 0; q < 2; ++q) {
        int idx  = q * 256 + tid;
        int srow = idx >> 2;
        int sseg = (idx & 3) * 8;
        aoff[q] = (size_t)(m0 + srow) * K_DIM + sseg;
        boff[q] = (size_t)(n0 + srow) * K_DIM + sseg;
        loff[q] = srow * LDAP + sseg;
    }

#if HAVE_ASYNC
    auto stage = [&](int buf, int kc) {
        const int kb = kc * KC;
        #pragma unroll
        for (int q = 0; q < 2; ++q) {
            async_cp16(XH + aoff[q] + kb, &ldsAhi[buf][loff[q]]);
            async_cp16(XL + aoff[q] + kb, &ldsAlo[buf][loff[q]]);
            async_cp16(WQ + boff[q] + kb, &ldsB  [buf][loff[q]]);
        }
    };
#else
    uint4 rha[2], rla[2], rbw[2];
    auto gload = [&](int kc) {
        const int kb = kc * KC;
        #pragma unroll
        for (int q = 0; q < 2; ++q) {
            rha[q] = *(const uint4*)(XH + aoff[q] + kb);
            rla[q] = *(const uint4*)(XL + aoff[q] + kb);
            rbw[q] = *(const uint4*)(WQ + boff[q] + kb);
        }
    };
    auto lstore = [&](int buf) {
        #pragma unroll
        for (int q = 0; q < 2; ++q) {
            *(uint4*)&ldsAhi[buf][loff[q]] = rha[q];
            *(uint4*)&ldsAlo[buf][loff[q]] = rla[q];
            *(uint4*)&ldsB  [buf][loff[q]] = rbw[q];
        }
    };
#endif

    v8f acc[2][4];
    #pragma unroll
    for (int ms = 0; ms < 2; ++ms)
        #pragma unroll
        for (int ns = 0; ns < 4; ++ns) { v8f z = {}; acc[ms][ns] = z; }

    const int akb = lg * 8;    // ISA 16-bit A fragment K base
    const int bkb = lg * 8;    // ISA 16-bit B fragment K base

    // hoisted fragment LDS offsets (buf 0)
    int aoffs[2], boffs[4];
    #pragma unroll
    for (int ms = 0; ms < 2; ++ms) aoffs[ms] = (wm * 32 + ms * 16 + l15) * LDAP + akb;
    #pragma unroll
    for (int ns = 0; ns < 4; ++ns) boffs[ns] = (wn * 64 + ns * 16 + l15) * LDAP + bkb;

    auto compute = [&](int buf) {
        const int bofs = buf * (MT * LDAP);
        v16bf ahi[2], alo[2], bfr[4];
        #pragma unroll
        for (int ms = 0; ms < 2; ++ms) {
            const __bf16* ph = &ldsAhi[0][bofs + aoffs[ms]];
            const __bf16* pl = &ldsAlo[0][bofs + aoffs[ms]];
            v8bf h0 = *(const v8bf*)(ph), h1 = *(const v8bf*)(ph + 16);
            v8bf l0 = *(const v8bf*)(pl), l1 = *(const v8bf*)(pl + 16);
            ahi[ms] = __builtin_shufflevector(h0, h1, 0,1,2,3,4,5,6,7,8,9,10,11,12,13,14,15);
            alo[ms] = __builtin_shufflevector(l0, l1, 0,1,2,3,4,5,6,7,8,9,10,11,12,13,14,15);
        }
        #pragma unroll
        for (int ns = 0; ns < 4; ++ns) {
            const __bf16* pb = &ldsB[0][bofs + boffs[ns]];
            v8bf b0 = *(const v8bf*)(pb), b1 = *(const v8bf*)(pb + 16);
            bfr[ns] = __builtin_shufflevector(b0, b1, 0,1,2,3,4,5,6,7,8,9,10,11,12,13,14,15);
        }
        #pragma unroll
        for (int ms = 0; ms < 2; ++ms)
            #pragma unroll
            for (int ns = 0; ns < 4; ++ns) {
                acc[ms][ns] = __builtin_amdgcn_wmma_f32_16x16x32_bf16(
                    false, ahi[ms], false, bfr[ns], (short)0, acc[ms][ns], false, false);
                acc[ms][ns] = __builtin_amdgcn_wmma_f32_16x16x32_bf16(
                    false, alo[ms], false, bfr[ns], (short)0, acc[ms][ns], false, false);
            }
    };

#if HAVE_ASYNC
    stage(0, 0);
    wait_async0();
    __syncthreads();
    for (int kc = 0; kc < NKC; ++kc) {
        if (kc + 1 < NKC) stage((kc + 1) & 1, kc + 1);   // async fill other buffer
        compute(kc & 1);
        if (kc + 1 < NKC) { wait_async0(); __syncthreads(); }
    }
#else
    gload(0);
    lstore(0);
    __syncthreads();
    for (int kc = 0; kc < NKC; ++kc) {
        if (kc + 1 < NKC) gload(kc + 1);
        compute(kc & 1);
        if (kc + 1 < NKC) {
            __syncthreads();
            lstore((kc + 1) & 1);
            __syncthreads();
        }
    }
#endif

    // C/D layout: VGPR r, lanes 0-15 (M=r,N=lane); lanes 16-31 (M=8+r,N=lane-16)
    #pragma unroll
    for (int ms = 0; ms < 2; ++ms) {
        int gm = m0 + wm * 32 + ms * 16 + lg * 8;
        #pragma unroll
        for (int ns = 0; ns < 4; ++ns) {
            int gn = n0 + wn * 64 + ns * 16 + l15;
            float bs = bsum[gn];
            #pragma unroll
            for (int r = 0; r < 8; ++r)
                out[(size_t)(gm + r) * N_DIM + gn] = acc[ms][ns][r] + bs;
        }
    }
}

// ------------------------------------------- fallback: fused fp32-in GEMM (r1)
__global__ __launch_bounds__(256) void splitfc_gemm_fused_kernel(
    const float* __restrict__ x, const float* __restrict__ W,
    const float* __restrict__ bsum, float* __restrict__ out) {

    __shared__ __bf16 ldsAhi[2][MT * LDAP];
    __shared__ __bf16 ldsAlo[2][MT * LDAP];
    __shared__ __bf16 ldsB  [2][NT * LDAP];

    const int tid  = (int)threadIdx.x;
    const int lane = tid & 31;
    const int wave = tid >> 5;
    const int wm   = wave >> 1;
    const int wn   = wave & 1;
    const int lg   = lane >> 4;
    const int l15  = lane & 15;
    const int m0   = (int)blockIdx.x * MT;
    const int n0   = (int)blockIdx.y * NT;

    float4 ra[4], rb[4];

    auto gload = [&](int kc) {
        const int kb = kc * KC;
        const int nb = kb >> 7;
        const int i0 = kb & 127;
        #pragma unroll
        for (int q = 0; q < 4; ++q) {
            int idx = q * 256 + tid;
            int row = idx >> 3;
            int c4  = (idx & 7) * 4;
            ra[q] = *(const float4*)(x + (size_t)(m0 + row) * K_DIM + kb + c4);
            rb[q] = *(const float4*)(W + ((size_t)nb * N_DIM + (n0 + row)) * IN_BLK + i0 + c4);
        }
    };

    auto lstore = [&](int buf) {
        #pragma unroll
        for (int q = 0; q < 4; ++q) {
            int idx = q * 256 + tid;
            int row = idx >> 3;
            int c4  = (idx & 7) * 4;
            int u2  = row * (LDAP >> 2) + (c4 >> 2);
            float f0 = ra[q].x, f1 = ra[q].y, f2 = ra[q].z, f3 = ra[q].w;
            unsigned h0 = __float_as_uint(f0) & 0xFFFF0000u;
            unsigned h1 = __float_as_uint(f1) & 0xFFFF0000u;
            unsigned h2 = __float_as_uint(f2) & 0xFFFF0000u;
            unsigned h3 = __float_as_uint(f3) & 0xFFFF0000u;
            float l0 = f0 - __uint_as_float(h0);
            float l1 = f1 - __uint_as_float(h1);
            float l2 = f2 - __uint_as_float(h2);
            float l3 = f3 - __uint_as_float(h3);
            uint2 hp, lp, bp;
            hp.x = (h0 >> 16) | h1;
            hp.y = (h2 >> 16) | h3;
            lp.x = bf16bits(l0) | (__float_as_uint(l1) & 0xFFFF0000u);
            lp.y = bf16bits(l2) | (__float_as_uint(l3) & 0xFFFF0000u);
            bp.x = bf16bits(fixed_quant_dev(rb[q].x)) |
                   (bf16bits(fixed_quant_dev(rb[q].y)) << 16);
            bp.y = bf16bits(fixed_quant_dev(rb[q].z)) |
                   (bf16bits(fixed_quant_dev(rb[q].w)) << 16);
            ((uint2*)&ldsAhi[buf][0])[u2] = hp;
            ((uint2*)&ldsAlo[buf][0])[u2] = lp;
            ((uint2*)&ldsB  [buf][0])[u2] = bp;
        }
    };

    v8f acc[2][4];
    #pragma unroll
    for (int ms = 0; ms < 2; ++ms)
        #pragma unroll
        for (int ns = 0; ns < 4; ++ns) { v8f z = {}; acc[ms][ns] = z; }

    const int akb = lg * 8;
    const int bkb = lg * 8;

    gload(0);
    lstore(0);
    __syncthreads();

    for (int kc = 0; kc < NKC; ++kc) {
        const int buf = kc & 1;
        if (kc + 1 < NKC) gload(kc + 1);

        v16bf ahi[2], alo[2], bfr[4];
        #pragma unroll
        for (int ms = 0; ms < 2; ++ms) {
            int mm = wm * 32 + ms * 16 + l15;
            const __bf16* ph = &ldsAhi[buf][mm * LDAP + akb];
            const __bf16* pl = &ldsAlo[buf][mm * LDAP + akb];
            v8bf h0 = *(const v8bf*)(ph), h1 = *(const v8bf*)(ph + 16);
            v8bf l0 = *(const v8bf*)(pl), l1 = *(const v8bf*)(pl + 16);
            ahi[ms] = __builtin_shufflevector(h0, h1, 0,1,2,3,4,5,6,7,8,9,10,11,12,13,14,15);
            alo[ms] = __builtin_shufflevector(l0, l1, 0,1,2,3,4,5,6,7,8,9,10,11,12,13,14,15);
        }
        #pragma unroll
        for (int ns = 0; ns < 4; ++ns) {
            int oo = wn * 64 + ns * 16 + l15;
            const __bf16* pb = &ldsB[buf][oo * LDAP + bkb];
            v8bf b0 = *(const v8bf*)(pb), b1 = *(const v8bf*)(pb + 16);
            bfr[ns] = __builtin_shufflevector(b0, b1, 0,1,2,3,4,5,6,7,8,9,10,11,12,13,14,15);
        }
        #pragma unroll
        for (int ms = 0; ms < 2; ++ms)
            #pragma unroll
            for (int ns = 0; ns < 4; ++ns) {
                acc[ms][ns] = __builtin_amdgcn_wmma_f32_16x16x32_bf16(
                    false, ahi[ms], false, bfr[ns], (short)0, acc[ms][ns], false, false);
                acc[ms][ns] = __builtin_amdgcn_wmma_f32_16x16x32_bf16(
                    false, alo[ms], false, bfr[ns], (short)0, acc[ms][ns], false, false);
            }

        if (kc + 1 < NKC) {
            __syncthreads();
            lstore((kc + 1) & 1);
            __syncthreads();
        }
    }

    #pragma unroll
    for (int ms = 0; ms < 2; ++ms) {
        int gm = m0 + wm * 32 + ms * 16 + lg * 8;
        #pragma unroll
        for (int ns = 0; ns < 4; ++ns) {
            int gn = n0 + wn * 64 + ns * 16 + l15;
            float bs = bsum[gn];
            #pragma unroll
            for (int r = 0; r < 8; ++r)
                out[(size_t)(gm + r) * N_DIM + gn] = acc[ms][ns][r] + bs;
        }
    }
}

extern "C" void kernel_launch(void* const* d_in, const int* in_sizes, int n_in,
                              void* d_out, int out_size, void* d_ws, size_t ws_size,
                              hipStream_t stream) {
    const float* x = (const float*)d_in[0];   // [1024, 4096]
    const float* W = (const float*)d_in[1];   // [32, 4096, 128]
    const float* b = (const float*)d_in[2];   // [32, 4096]
    float* out = (float*)d_out;               // [1024, 4096]

    char* ws = (char*)d_ws;
    float* bsum = (float*)ws;                                  // 16 KB
    const size_t WQ_BYTES = (size_t)N_DIM * K_DIM * 2;         // 32 MB
    const size_t XA_BYTES = (size_t)M_DIM * K_DIM * 2;         // 8 MB each
    const size_t NEED = 16384 + WQ_BYTES + 2 * XA_BYTES;       // ~50.3 MB

    bias_sum_kernel<<<N_DIM / 256, 256, 0, stream>>>(b, bsum);

    dim3 grid(M_DIM / MT, N_DIM / NT);        // (8, 32)

    if (ws_size >= NEED) {
        __bf16* WQ = (__bf16*)(ws + 16384);
        __bf16* XH = (__bf16*)(ws + 16384 + WQ_BYTES);
        __bf16* XL = XH + (size_t)M_DIM * K_DIM;
        quantW_kernel<<<(N_DIM * (K_DIM / 8)) / 256, 256, 0, stream>>>(W, WQ);
        splitX_kernel<<<(M_DIM * (K_DIM / 8)) / 256, 256, 0, stream>>>(x, XH, XL);
        splitfc_gemm_pre_kernel<<<grid, 256, 0, stream>>>(XH, XL, WQ, bsum, out);
    } else {
        splitfc_gemm_fused_kernel<<<grid, 256, 0, stream>>>(x, W, bsum, out);
    }
}
